// UPFusion_3882650436787
// MI455X (gfx1250) — matrix operand order repaired and checked by reading, not compiled, not run on hardware
//
#include <hip/hip_runtime.h>
#include <math.h>

// ---------------------------------------------------------------------------
// Problem constants (from reference setup_inputs)
// ---------------------------------------------------------------------------
#define Bn   4
#define Cn   128          // base channel count
#define Hn   128
#define Wn   128
#define HWn  16384        // Hn*Wn
#define HEADS 8
#define CH   16           // channels per head

#define TILE_N    64      // columns per block tile
#define TILE_K    32      // K step (one bf16 WMMA depth)
#define LDS_PITCH 40      // bf16 elements per column (80 B, bank-conflict pad)

typedef __attribute__((ext_vector_type(16))) __bf16 v16bf;
typedef __attribute__((ext_vector_type(8)))  float  v8f;

// ---------------------------------------------------------------------------
// WMMA GEMM:  Y[b][o][n] = sum_k W[o][k] * X[b][map(k)][n] (+bias)(+res)
// Block = 256 threads (8 waves). Block tile: 128(M) x 64(N); wave tile 16x64.
// X tile staged to LDS transposed [col][k] in bf16 so B fragments are two
// contiguous ds_load_b128 per lane. K must be a multiple of 32 (callers pad).
// N is fixed at HWn (16384) so row offsets are shifts, not 64-bit multiplies.
// HAS_MAP is a template flag so the staging loop has no per-iteration branch.
// Fragment k order is contiguous, k = 16*(lane/16) + s, for both A and B (wave32).
// ---------------------------------------------------------------------------
template <bool HAS_MAP>
__global__ __launch_bounds__(256)
void gemm_wmma_k(const float* __restrict__ W, int O, int K,
                 const float* __restrict__ bias,
                 const float* __restrict__ X, long xbs,
                 const int* __restrict__ map, int mapStride,
                 const float* __restrict__ res, long rbs,
                 float* __restrict__ Y, long ybs)
{
    __shared__ __align__(16) __bf16 T[TILE_N * LDS_PITCH];

    const int t    = threadIdx.x;
    const int wave = t >> 5;
    const int lane = t & 31;
    const int hi   = lane >> 4;          // lane half
    const int l15  = lane & 15;
    const int b    = blockIdx.z;
    const int n0   = blockIdx.x * TILE_N;
    const int m0   = blockIdx.y * 128 + wave * 16;

    const float* Xb = X + (long)b * xbs;
    const int*   mp = HAS_MAP ? (map + b * mapStride) : nullptr;

    // A row for this lane (clamped so loads are unconditional; stores guarded)
    const int arow = (m0 + l15 < O) ? (m0 + l15) : (O - 1);
    const float* wr = W + (long)arow * K;

    // staging role: thread -> (column, k-subgroup of 8)
    const int scol = t & 63;
    const int skg  = t >> 6;             // 0..3

    v8f acc0 = {}, acc1 = {}, acc2 = {}, acc3 = {};

    for (int k0 = 0; k0 < K; k0 += TILE_K) {
        // ---- stage 32K x 64N f32 -> LDS bf16 transposed ------------------
        {
            int rows[8];
            if (HAS_MAP) {  // batched map fetch: 2 x b128, then 8 indep loads
                const int4 r0 = *(const int4*)(mp + k0 + skg * 8);
                const int4 r1 = *(const int4*)(mp + k0 + skg * 8 + 4);
                rows[0] = r0.x; rows[1] = r0.y; rows[2] = r0.z; rows[3] = r0.w;
                rows[4] = r1.x; rows[5] = r1.y; rows[6] = r1.z; rows[7] = r1.w;
            } else {
#pragma unroll
                for (int j = 0; j < 8; ++j) rows[j] = k0 + skg * 8 + j;
            }
            const float* xc = Xb + (n0 + scol);
            float v[8];
#pragma unroll
            for (int j = 0; j < 8; ++j) v[j] = xc[(long)rows[j] * HWn];
            __bf16 pk[8];
#pragma unroll
            for (int j = 0; j < 8; ++j) pk[j] = (__bf16)v[j];
            __builtin_memcpy(&T[(size_t)scol * LDS_PITCH + skg * 8], pk, 16);
        }
        __syncthreads();

        // ---- A fragment: 4 x b128 from W (K-contiguous), pack bf16 -------
        v16bf afrag;
        {
            const float4 w0 = *(const float4*)(wr + k0 + hi * 16);
            const float4 w1 = *(const float4*)(wr + k0 + hi * 16 + 4);
            const float4 w2 = *(const float4*)(wr + k0 + hi * 16 + 8);
            const float4 w3 = *(const float4*)(wr + k0 + hi * 16 + 12);
            afrag[0]  = (__bf16)w0.x; afrag[1]  = (__bf16)w0.y;
            afrag[2]  = (__bf16)w0.z; afrag[3]  = (__bf16)w0.w;
            afrag[4]  = (__bf16)w1.x; afrag[5]  = (__bf16)w1.y;
            afrag[6]  = (__bf16)w1.z; afrag[7]  = (__bf16)w1.w;
            afrag[8]  = (__bf16)w2.x; afrag[9]  = (__bf16)w2.y;
            afrag[10] = (__bf16)w2.z; afrag[11] = (__bf16)w2.w;
            afrag[12] = (__bf16)w3.x; afrag[13] = (__bf16)w3.y;
            afrag[14] = (__bf16)w3.z; afrag[15] = (__bf16)w3.w;
        }

        // ---- B fragments from LDS (16 contiguous bf16 per lane) ----------
        v16bf bf0, bf1, bf2, bf3;
        __builtin_memcpy(&bf0, &T[(size_t)( 0 + l15) * LDS_PITCH + hi * 16], 32);
        __builtin_memcpy(&bf1, &T[(size_t)(16 + l15) * LDS_PITCH + hi * 16], 32);
        __builtin_memcpy(&bf2, &T[(size_t)(32 + l15) * LDS_PITCH + hi * 16], 32);
        __builtin_memcpy(&bf3, &T[(size_t)(48 + l15) * LDS_PITCH + hi * 16], 32);

        acc0 = __builtin_amdgcn_wmma_f32_16x16x32_bf16(false, afrag, false, bf0, (short)0, acc0, false, false);
        acc1 = __builtin_amdgcn_wmma_f32_16x16x32_bf16(false, afrag, false, bf1, (short)0, acc1, false, false);
        acc2 = __builtin_amdgcn_wmma_f32_16x16x32_bf16(false, afrag, false, bf2, (short)0, acc2, false, false);
        acc3 = __builtin_amdgcn_wmma_f32_16x16x32_bf16(false, afrag, false, bf3, (short)0, acc3, false, false);
        __syncthreads();
    }

    // ---- epilogue: D layout VGPR r -> M = mbase+r, col per lane ----------
    float* Yb = Y + (long)b * ybs;
    const float* Rb = res ? (res + (long)b * rbs) : nullptr;
    const int mbase = m0 + hi * 8;
#pragma unroll
    for (int tt = 0; tt < 4; ++tt) {
        const v8f acc = (tt == 0) ? acc0 : (tt == 1) ? acc1 : (tt == 2) ? acc2 : acc3;
        const int col = n0 + tt * 16 + l15;
#pragma unroll
        for (int r = 0; r < 8; ++r) {
            const int m = mbase + r;
            if (m < O) {
                float v = acc[r];
                if (bias) v += bias[m];
                if (Rb)   v += Rb[(long)m * HWn + col];
                Yb[(long)m * HWn + col] = v;
            }
        }
    }
}

// ---------------------------------------------------------------------------
// concat[b][c][n]: c<128 -> vi, else ir
// ---------------------------------------------------------------------------
__global__ __launch_bounds__(256)
void concat_k(const float* __restrict__ vi, const float* __restrict__ ir,
              float* __restrict__ out)
{
    long id  = (long)blockIdx.x * 256 + threadIdx.x;
    long per = (long)Cn * HWn;
    long b   = id / (2 * per);
    long r   = id - b * 2 * per;
    out[id]  = (r < per) ? vi[b * per + r] : ir[b * per + (r - per)];
}

// ---------------------------------------------------------------------------
// per-(b,channel) mean over HW
// ---------------------------------------------------------------------------
__global__ __launch_bounds__(256)
void squeeze_k(const float* __restrict__ X, float* __restrict__ sq)
{
    const int bc = blockIdx.x;
    const float* xp = X + (long)bc * HWn;
    float s = 0.0f;
    for (int i = threadIdx.x; i < HWn; i += 256) s += xp[i];
    __shared__ float red[256];
    red[threadIdx.x] = s;
    __syncthreads();
    for (int k = 128; k > 0; k >>= 1) {
        if (threadIdx.x < k) red[threadIdx.x] += red[threadIdx.x + k];
        __syncthreads();
    }
    if (threadIdx.x == 0) sq[bc] = red[0] * (1.0f / HWn);
}

// ---------------------------------------------------------------------------
// SE MLP + top_k(cw,128) indices ; text MLP + top_k(tf,256) (full argsort).
// One block per batch, 256 threads. Tie-break: strict '>' -> lowest index.
// ---------------------------------------------------------------------------
__global__ __launch_bounds__(256)
void se_text_k(const float* __restrict__ sq,
               const float* __restrict__ se_w1, const float* __restrict__ se_w2,
               const float* __restrict__ text,
               const float* __restrict__ tw1, const float* __restrict__ tb1,
               const float* __restrict__ tw2, const float* __restrict__ tb2,
               int* __restrict__ idx_out, int* __restrict__ soft_out)
{
    const int b = blockIdx.x;
    const int t = threadIdx.x;
    __shared__ float h1[32];
    __shared__ float cw[256];
    __shared__ float th[512];
    __shared__ float tf[256];

    if (t < 32) {
        float s = 0.0f;
        for (int c = 0; c < 256; ++c) s += sq[b * 256 + c] * se_w1[t * 256 + c];
        h1[t] = fmaxf(s, 0.0f);
    }
    __syncthreads();
    {
        float s = 0.0f;
        for (int j = 0; j < 32; ++j) s += h1[j] * se_w2[t * 32 + j];
        cw[t] = 1.0f / (1.0f + expf(-s));
    }
    for (int j = t; j < 512; j += 256) {
        float s = tb1[j];
        for (int c = 0; c < 512; ++c) s += text[b * 512 + c] * tw1[j * 512 + c];
        th[j] = fmaxf(s, 0.0f);
    }
    __syncthreads();
    {
        float s = tb2[t];
        for (int j = 0; j < 512; ++j) s += th[j] * tw2[t * 512 + j];
        tf[t] = s;
    }
    __syncthreads();
    if (t == 0) {
        for (int r = 0; r < 128; ++r) {
            int bi = 0; float bv = cw[0];
            for (int j = 1; j < 256; ++j) if (cw[j] > bv) { bv = cw[j]; bi = j; }
            idx_out[b * 128 + r] = bi;
            cw[bi] = -3.0e38f;
        }
        for (int r = 0; r < 256; ++r) {
            int bi = 0; float bv = tf[0];
            for (int j = 1; j < 256; ++j) if (tf[j] > bv) { bv = tf[j]; bi = j; }
            soft_out[b * 256 + r] = bi;
            tf[bi] = -3.0e38f;
        }
    }
}

// ---------------------------------------------------------------------------
// LayerNorm over C=128 at each (b,n)
// ---------------------------------------------------------------------------
__global__ __launch_bounds__(256)
void ln_k(const float* __restrict__ X, const float* __restrict__ w,
          const float* __restrict__ bia, float* __restrict__ Y)
{
    const int b = blockIdx.y;
    const int n = blockIdx.x * 256 + threadIdx.x;
    const float* xp = X + (long)b * Cn * HWn + n;
    float s = 0.0f, s2 = 0.0f;
    for (int c = 0; c < Cn; ++c) {
        float v = xp[(long)c * HWn];
        s += v; s2 += v * v;
    }
    const float mu  = s * (1.0f / Cn);
    const float var = s2 * (1.0f / Cn) - mu * mu;
    const float rs  = rsqrtf(var + 1e-5f);
    float* yp = Y + (long)b * Cn * HWn + n;
    for (int c = 0; c < Cn; ++c)
        yp[(long)c * HWn] = (xp[(long)c * HWn] - mu) * rs * w[c] + bia[c];
}

// ---------------------------------------------------------------------------
// Depthwise 3x3, SAME zero padding. X,Y: [B][C][H][W]; W3: [C][3][3]
// ---------------------------------------------------------------------------
__global__ __launch_bounds__(256)
void dwconv_k(const float* __restrict__ X, const float* __restrict__ W3,
              float* __restrict__ Y, int C)
{
    long id = (long)blockIdx.x * 256 + threadIdx.x;
    long total = (long)Bn * C * HWn;
    if (id >= total) return;
    int  pix = (int)(id % HWn);
    long bc  = id / HWn;
    int  c   = (int)(bc % C);
    int  hh  = pix >> 7;
    int  ww  = pix & (Wn - 1);
    const float* xp = X + bc * HWn;
    const float* wp = W3 + c * 9;
    float acc = 0.0f;
#pragma unroll
    for (int dy = -1; dy <= 1; ++dy) {
        int y = hh + dy;
        if ((unsigned)y >= Hn) continue;
#pragma unroll
        for (int dx = -1; dx <= 1; ++dx) {
            int x = ww + dx;
            if ((unsigned)x >= Wn) continue;
            acc += xp[y * Wn + x] * wp[(dy + 1) * 3 + (dx + 1)];
        }
    }
    Y[id] = acc;
}

// ---------------------------------------------------------------------------
// per-(b,c) reciprocal L2 norm over HW; row = X + b*bstride + c*HW
// ---------------------------------------------------------------------------
__global__ __launch_bounds__(256)
void norms_k(const float* __restrict__ X, long bstride, int C,
             float* __restrict__ rn)
{
    const int bc = blockIdx.x;
    const int b  = bc / C;
    const int c  = bc % C;
    const float* xp = X + (long)b * bstride + (long)c * HWn;
    float s = 0.0f;
    for (int i = threadIdx.x; i < HWn; i += 256) { float v = xp[i]; s += v * v; }
    __shared__ float red[256];
    red[threadIdx.x] = s;
    __syncthreads();
    for (int k = 128; k > 0; k >>= 1) {
        if (threadIdx.x < k) red[threadIdx.x] += red[threadIdx.x + k];
        __syncthreads();
    }
    if (threadIdx.x == 0) rn[bc] = 1.0f / fmaxf(sqrtf(red[0]), 1e-12f);
}

// ---------------------------------------------------------------------------
// Gram (q.k^T per head), scale by rnorms*temperature, top-14 threshold mask,
// softmax -> score[b][h][16][16]. One block per (b,h); thread = (c,d) pair.
// ---------------------------------------------------------------------------
__global__ __launch_bounds__(256)
void gram_score_k(const float* __restrict__ Q, const float* __restrict__ Kv,
                  const float* __restrict__ rq, const float* __restrict__ rk,
                  const float* __restrict__ temp, float* __restrict__ score)
{
    const int bh = blockIdx.x;
    const int b  = bh >> 3;
    const int h  = bh & 7;
    const int t  = threadIdx.x;
    const int c  = t >> 4;
    const int d  = t & 15;
    const float* qp = Q  + ((long)(b * 128 + h * CH)) * HWn;
    const float* kp = Kv + ((long)(b * 256 + h * CH)) * HWn;

    __shared__ float qt[16 * 129];
    __shared__ float kt[16 * 129];
    float acc = 0.0f;

    for (int n0 = 0; n0 < HWn; n0 += 128) {
#pragma unroll
        for (int i = 0; i < 8; ++i) {
            int e  = t + i * 256;
            int cc = e >> 7;
            int nn = e & 127;
            qt[cc * 129 + nn] = qp[(long)cc * HWn + n0 + nn];
            kt[cc * 129 + nn] = kp[(long)cc * HWn + n0 + nn];
        }
        __syncthreads();
#pragma unroll 8
        for (int j = 0; j < 128; ++j) acc += qt[c * 129 + j] * kt[d * 129 + j];
        __syncthreads();
    }

    const float a = acc * rq[b * 128 + h * CH + c] * rk[b * 128 + h * CH + d] * temp[h];
    __shared__ float g[256];
    g[c * 16 + d] = a;
    __syncthreads();

    if (t < 16) {  // thread t owns row t
        float row[16], tmp[16];
#pragma unroll
        for (int j = 0; j < 16; ++j) { row[j] = g[t * 16 + j]; tmp[j] = row[j]; }
        // threshold = 14th largest (kk = int(16*0.9) = 14)
        float thr = 0.0f;
        for (int r = 0; r < 14; ++r) {
            int bi = 0; float bv = tmp[0];
            for (int j = 1; j < 16; ++j) if (tmp[j] > bv) { bv = tmp[j]; bi = j; }
            thr = bv;
            tmp[bi] = -3.0e38f;
        }
        float mx = row[0];
        for (int j = 1; j < 16; ++j) mx = fmaxf(mx, row[j]);
        float e[16];
        float s = 0.0f;
        for (int j = 0; j < 16; ++j) {
            e[j] = (row[j] >= thr) ? expf(row[j] - mx) : 0.0f;
            s += e[j];
        }
        const float inv = 1.0f / s;
        for (int j = 0; j < 16; ++j)
            score[(long)bh * 256 + t * 16 + j] = e[j] * inv;
    }
}

// ---------------------------------------------------------------------------
// out[b][h*16+c][n] = attn4 * sum_d score[b][h][c][d] * v[b][h*16+d][n]
// v = kv channels [128..256)
// ---------------------------------------------------------------------------
__global__ __launch_bounds__(256)
void attn_out_k(const float* __restrict__ score, const float* __restrict__ kvbuf,
                const float* __restrict__ attn4, float* __restrict__ out)
{
    const int bh = blockIdx.x;
    const int b  = bh >> 3;
    const int h  = bh & 7;
    const int n  = blockIdx.y * 256 + threadIdx.x;
    __shared__ float s[256];
    s[threadIdx.x] = score[(long)bh * 256 + threadIdx.x];
    __syncthreads();
    const float a4 = attn4[0];
    float v[16];
#pragma unroll
    for (int d = 0; d < 16; ++d)
        v[d] = kvbuf[((long)(b * 256 + 128 + h * CH + d)) * HWn + n];
#pragma unroll
    for (int c = 0; c < 16; ++c) {
        float o = 0.0f;
#pragma unroll
        for (int d = 0; d < 16; ++d) o += s[c * 16 + d] * v[d];
        out[((long)(b * 128 + h * CH + c)) * HWn + n] = o * a4;
    }
}

// ---------------------------------------------------------------------------
// GeGLU, zero-padded to 352 channels so the following GEMM needs no K guard:
// G[b][j][n] = j<340 ? gelu_exact(Y[b][j][n]) * Y[b][340+j][n] : 0
// ---------------------------------------------------------------------------
__global__ __launch_bounds__(256)
void geglu_k(const float* __restrict__ Y2, float* __restrict__ G)
{
    long id = (long)blockIdx.x * 256 + threadIdx.x;
    const long perp = (long)352 * HWn;
    if (id >= (long)Bn * perp) return;
    long b = id / perp;
    long r = id - b * perp;
    long j = r / HWn;
    float o = 0.0f;
    if (j < 340) {
        long n  = r - j * HWn;
        float x1 = Y2[b * 680 * (long)HWn + j * HWn + n];
        float x2 = Y2[b * 680 * (long)HWn + (340 + j) * HWn + n];
        o = 0.5f * x1 * (1.0f + erff(x1 * 0.70710678118654752f)) * x2;
    }
    G[id] = o;
}

// ---------------------------------------------------------------------------
// Pad ffn_out_w [128,340] -> [128,352] with zeros (K multiple of 32)
// ---------------------------------------------------------------------------
__global__ __launch_bounds__(256)
void padw_k(const float* __restrict__ src, float* __restrict__ dst)
{
    int id = blockIdx.x * 256 + threadIdx.x;
    if (id >= 128 * 352) return;
    int o = id / 352;
    int k = id - o * 352;
    dst[id] = (k < 340) ? src[o * 340 + k] : 0.0f;
}

// ---------------------------------------------------------------------------
// Host-side orchestration
// ---------------------------------------------------------------------------
extern "C" void kernel_launch(void* const* d_in, const int* in_sizes, int n_in,
                              void* d_out, int out_size, void* d_ws, size_t ws_size,
                              hipStream_t stream)
{
    (void)in_sizes; (void)n_in; (void)out_size; (void)ws_size;

    const float* vi        = (const float*)d_in[0];
    const float* ir        = (const float*)d_in[1];
    const float* text_code = (const float*)d_in[2];
    const float* se_w1     = (const float*)d_in[3];
    const float* se_w2     = (const float*)d_in[4];
    const float* c1_w      = (const float*)d_in[5];
    const float* c1_b      = (const float*)d_in[6];
    const float* co_w      = (const float*)d_in[7];
    const float* co_b      = (const float*)d_in[8];
    const float* tw1       = (const float*)d_in[9];
    const float* tb1       = (const float*)d_in[10];
    const float* tw2       = (const float*)d_in[11];
    const float* tb2       = (const float*)d_in[12];
    const float* n1w       = (const float*)d_in[13];
    const float* n1b       = (const float*)d_in[14];
    const float* n2w       = (const float*)d_in[15];
    const float* n2b       = (const float*)d_in[16];
    const float* n3w       = (const float*)d_in[17];
    const float* n3b       = (const float*)d_in[18];
    const float* q_w       = (const float*)d_in[19];
    const float* q_dw      = (const float*)d_in[20];
    const float* kv_w      = (const float*)d_in[21];
    const float* kv_dw     = (const float*)d_in[22];
    const float* temp      = (const float*)d_in[23];
    const float* attn4     = (const float*)d_in[24];
    const float* proj_w    = (const float*)d_in[25];
    const float* ffn_in_w  = (const float*)d_in[26];
    const float* ffn_dw    = (const float*)d_in[27];
    const float* ffn_out_w = (const float*)d_in[28];

    float* out0 = (float*)d_out;                 // output  [4,128,128,128]
    float* out1 = out0 + (long)Bn * Cn * HWn;    // img_feature2

    // ---- workspace layout (floats) ---------------------------------------
    float* ws = (float*)d_ws;
    const long SZ1 = (long)Bn * Cn * HWn;        //  8,388,608 (128 ch)
    const long SZ2 = 2 * SZ1;                    // 16,777,216 (256 ch)
    float* A  = ws;                 // concat           [4,256,HW]
    float* Bf = A  + SZ2;           // img              [4,256,HW]
    float* Cf = Bf + SZ2;           // q_in             [4,128,HW]
    float* Df = Cf + SZ1;           // ln1
    float* Ef = Df + SZ1;           // ln2
    float* Ff = Ef + SZ1;           // q (pre-dw)
    float* Gf = Ff + SZ1;           // kv (pre-dw)     [4,256,HW]
    float* Hf = Gf + SZ2;           // q
    float* If = Hf + SZ1;           // kv              [4,256,HW]
    float* Jf = If + SZ2;           // attention out (pre-proj)
    float* Kf = Jf + SZ1;           // att  (outside reuse region)
    float* Lf = Kf + SZ1;           // ln3  (outside reuse region)
    float* smallf = Lf + SZ1;
    float* sq     = smallf;               // [4,256]
    float* rq     = sq + 1024;            // [4,128]
    float* rk     = rq + 512;             // [4,128]
    float* score  = rk + 512;             // [4,8,16,16]
    float* wpad   = score + 8192;         // [128,352]
    int*   idx    = (int*)(wpad + 128 * 352); // [4,128]  (16B aligned)
    int*   soft   = idx + 512;                // [4,256]  (16B aligned)
    // FFN reuses A..Jf region (dead by then): (680+680+352) ch fit there
    float* M1 = A;                    // ffn pre-dw   [4,680,HW]
    float* N2 = A + 44564480;         // ffn post-dw  [4,680,HW]
    float* Pg = A + 89128960;         // gelu product [4,352,HW] (zero-padded)

    auto gemm = [&](const float* Wm, int O, int K, const float* bias,
                    const float* X, long xbs, const int* map, int mapStride,
                    const float* res, long rbs, float* Y, long ybs) {
        dim3 g(HWn / TILE_N, (O + 127) / 128, Bn);
        if (map)
            gemm_wmma_k<true><<<g, 256, 0, stream>>>(Wm, O, K, bias, X, xbs,
                                                     map, mapStride, res, rbs, Y, ybs);
        else
            gemm_wmma_k<false><<<g, 256, 0, stream>>>(Wm, O, K, bias, X, xbs,
                                                      nullptr, 0, res, rbs, Y, ybs);
    };

    // 1. concat
    concat_k<<<(int)(SZ2 / 256), 256, 0, stream>>>(vi, ir, A);
    // 2. squeeze means + SE/text MLPs + top-k index sorts; weight padding
    squeeze_k<<<Bn * 256, 256, 0, stream>>>(A, sq);
    se_text_k<<<Bn, 256, 0, stream>>>(sq, se_w1, se_w2, text_code,
                                      tw1, tb1, tw2, tb2, idx, soft);
    padw_k<<<(128 * 352 + 255) / 256, 256, 0, stream>>>(ffn_out_w, wpad);
    // 3. img = conv1x1(gather(concat, idx))          [O=256, K=128]
    gemm(c1_w, 256, 128, c1_b, A, (long)256 * HWn, idx, 128,
         nullptr, 0, Bf, (long)256 * HWn);
    // 4. img_feature2 = conv_out(concat)             [O=128, K=256] -> d_out[1]
    gemm(co_w, 128, 256, co_b, A, (long)256 * HWn, nullptr, 0,
         nullptr, 0, out1, (long)128 * HWn);
    // 5. q_in = conv_out(gather(img, soft_idx))      [O=128, K=256]
    gemm(co_w, 128, 256, co_b, Bf, (long)256 * HWn, soft, 256,
         nullptr, 0, Cf, (long)128 * HWn);
    // 6. layernorms
    ln_k<<<dim3(HWn / 256, Bn), 256, 0, stream>>>(Cf,   n1w, n1b, Df);
    ln_k<<<dim3(HWn / 256, Bn), 256, 0, stream>>>(out1, n2w, n2b, Ef);
    // 7. q / kv projections + depthwise 3x3
    gemm(q_w, 128, 128, nullptr, Df, (long)128 * HWn, nullptr, 0,
         nullptr, 0, Ff, (long)128 * HWn);
    gemm(kv_w, 256, 128, nullptr, Ef, (long)128 * HWn, nullptr, 0,
         nullptr, 0, Gf, (long)256 * HWn);
    dwconv_k<<<(int)(((long)Bn * 128 * HWn + 255) / 256), 256, 0, stream>>>(Ff, q_dw,  Hf, 128);
    dwconv_k<<<(int)(((long)Bn * 256 * HWn + 255) / 256), 256, 0, stream>>>(Gf, kv_dw, If, 256);
    // 8. channel attention (l2-normed gram, top-k mask, softmax, score.V)
    norms_k<<<Bn * 128, 256, 0, stream>>>(Hf, (long)128 * HWn, 128, rq);
    norms_k<<<Bn * 128, 256, 0, stream>>>(If, (long)256 * HWn, 128, rk);
    gram_score_k<<<Bn * HEADS, 256, 0, stream>>>(Hf, If, rq, rk, temp, score);
    attn_out_k<<<dim3(Bn * HEADS, HWn / 256), 256, 0, stream>>>(score, If, attn4, Jf);
    // 9. att = proj(attn_out)
    gemm(proj_w, 128, 128, nullptr, Jf, (long)128 * HWn, nullptr, 0,
         nullptr, 0, Kf, (long)128 * HWn);
    // 10. FFN: ln3 -> ffn_in -> dw -> geglu(pad 352) -> ffn_out(+att) -> d_out[0]
    ln_k<<<dim3(HWn / 256, Bn), 256, 0, stream>>>(Kf, n3w, n3b, Lf);
    gemm(ffn_in_w, 680, 128, nullptr, Lf, (long)128 * HWn, nullptr, 0,
         nullptr, 0, M1, (long)680 * HWn);
    dwconv_k<<<(int)(((long)Bn * 680 * HWn + 255) / 256), 256, 0, stream>>>(M1, ffn_dw, N2, 680);
    geglu_k<<<(int)(((long)Bn * 352 * HWn + 255) / 256), 256, 0, stream>>>(N2, Pg);
    gemm(wpad, 128, 352, nullptr, Pg, (long)352 * HWn, nullptr, 0,
         Kf, (long)128 * HWn, out0, (long)128 * HWn);
}
